// ResBlock_14070312862110
// MI455X (gfx1250) — matrix operand order repaired and checked by reading, not compiled, not run on hardware
//
#include <hip/hip_runtime.h>

// ---------------------------------------------------------------------------
// CDNA5 (gfx1250) submanifold sparse-conv ResBlock via V_WMMA_F32_16X16X32_F16
// ---------------------------------------------------------------------------

typedef _Float16 v8h  __attribute__((ext_vector_type(8)));
typedef _Float16 v16h __attribute__((ext_vector_type(16)));
typedef float    v8f  __attribute__((ext_vector_type(8)));

#define NVOX   150000
#define NCOUT  64
#define BN_EPS 1e-3f
#define SLOPE  0.01f
#define TILES  (NVOX / 16)                       // 9375 (exact)
#define WPB    8                                 // waves per block (256 threads)
#define NBLK   ((TILES + WPB - 1) / WPB)         // 1172

// ---------------------------------------------------------------------------
// feats f32 [N,32] -> f16 [(N+1),32], row N zeroed (gather sentinel)
// ---------------------------------------------------------------------------
__global__ __launch_bounds__(256)
void prep_feats(const float* __restrict__ f, _Float16* __restrict__ fh) {
    size_t i = ((size_t)blockIdx.x * 256 + threadIdx.x) * 8;
    const size_t total = (size_t)(NVOX + 1) * 32;
    if (i >= total) return;
    v8h y = {};
    if (i < (size_t)NVOX * 32) {
        const float* s = f + i;
        #pragma unroll
        for (int e = 0; e < 8; ++e) y[e] = (_Float16)s[e];
    }
    *(v8h*)(fh + i) = y;
}

// ---------------------------------------------------------------------------
// W f32 [9][Cin][64] -> Wt f16 [9][64][Cin]  (B-fragment friendly layout)
// ---------------------------------------------------------------------------
__global__ __launch_bounds__(256)
void prep_w(const float* __restrict__ W, _Float16* __restrict__ Wt, int Cin) {
    int e = blockIdx.x * 256 + threadIdx.x;
    int total = 9 * NCOUT * Cin;
    if (e >= total) return;
    int k  = e / (NCOUT * Cin);
    int r  = e - k * (NCOUT * Cin);
    int co = r / Cin;
    int ci = r - co * Cin;
    Wt[e] = (_Float16)W[((size_t)k * Cin + ci) * NCOUT + co];
}

// ---------------------------------------------------------------------------
// Core: gather-GEMM (9-pt stencil) + LeakyReLU + per-block BN partial stats.
// One wave = 16 voxels x 64 cout (4x v8f accumulators).
// A k order chosen here (contiguous, not the ISA table order): lane<16 -> K 0..15 ; lane>=16 -> K 16..31
// B k order chosen here (same contiguous order as A): lane<16 -> K 0..15 of col=lane ; lane>=16 -> K 16..31
// Epilogue transposes the 16x64 tile through per-wave LDS so global stores
// are fully-coalesced b128 ops instead of 2-byte scatters.
// ---------------------------------------------------------------------------
template<int CIN>
__global__ __launch_bounds__(256)
void subm_conv_wmma(const _Float16* __restrict__ in,    // [(N+1)*CIN], row N = 0
                    const int*      __restrict__ nbr,   // [9][N], sentinel == N
                    const _Float16* __restrict__ Wt,    // [9][64][CIN] f16
                    _Float16*       __restrict__ outF,  // [N][64] post-lrelu, pre-BN
                    float*          __restrict__ partials) // [2][NBLK][64]
{
    __shared__ float psum[NCOUT];
    __shared__ float psumsq[NCOUT];
    __shared__ _Float16 xpose[WPB][16][NCOUT];           // 16KB: per-wave staging
    const int tid = threadIdx.x;
    if (tid < NCOUT) { psum[tid] = 0.f; psumsq[tid] = 0.f; }
    __syncthreads();

    const int wave = tid >> 5;
    const int lane = tid & 31;
    const int col  = lane & 15;     // A row / B,D column within 16-tile
    const int hsel = lane >> 4;
    const int tile = blockIdx.x * WPB + wave;
    const bool wave_valid = (tile < TILES);   // wave-uniform: EXEC all-1s inside

    v8f acc[4] = {{0.f}, {0.f}, {0.f}, {0.f}};

    if (wave_valid) {
        const int base = tile * 16;
        for (int k = 0; k < 9; ++k) {
            const int idx = nbr[k * NVOX + base + col];   // idx==NVOX -> zero row
            const _Float16* fp = in + (size_t)idx * CIN;
            const _Float16* wk = Wt + (size_t)k * NCOUT * CIN + hsel * 16;
            #pragma unroll
            for (int ks = 0; ks < CIN / 32; ++ks) {
                const int kb = ks * 32 + hsel * 16;
                v8h lo = *(const v8h*)(fp + kb);          // K kb..kb+7
                v8h hi = *(const v8h*)(fp + kb + 8);      // K kb+8..kb+15
                v16h a = __builtin_shufflevector(lo, hi,
                         0,1,2,3,4,5,6,7,8,9,10,11,12,13,14,15);
                const _Float16* wkk = wk + ks * 32;
                #pragma unroll
                for (int t = 0; t < 4; ++t) {
                    v16h b = *(const v16h*)(wkk + (size_t)(col + t * 16) * CIN);
                    acc[t] = __builtin_amdgcn_wmma_f32_16x16x32_f16(
                                 false, a, false, b, (short)0, acc[t],
                                 false, false);
                }
            }
        }
        // Epilogue: LeakyReLU + BN partial sums (ds_add_f32) + LDS transpose.
        // D layout: lane holds column `col`, rows M = hsel*8 + j (j=0..7).
        #pragma unroll
        for (int t = 0; t < 4; ++t) {
            const int ch = t * 16 + col;
            float s = 0.f, q = 0.f;
            #pragma unroll
            for (int j = 0; j < 8; ++j) {
                float x = acc[t][j];
                x = (x >= 0.f) ? x : SLOPE * x;
                s += x; q += x * x;
                xpose[wave][hsel * 8 + j][ch] = (_Float16)x;
            }
            atomicAdd(&psum[ch], s);      // ds_add_f32
            atomicAdd(&psumsq[ch], q);
        }
        // Coalesced writeout: 16 rows x 128B = 128 chunks of 8 halfs.
        // LDS ops of one wave are in-order -> no barrier for wave-private tile.
        const _Float16* src = &xpose[wave][0][0];
        _Float16* dst = outF + (size_t)base * NCOUT;
        #pragma unroll
        for (int c = 0; c < 4; ++c) {
            const int off8 = (lane + 32 * c) * 8;
            *(v8h*)(dst + off8) = *(const v8h*)(src + off8);
        }
    }
    __syncthreads();
    if (tid < NCOUT) {
        partials[(size_t)blockIdx.x * NCOUT + tid]          = psum[tid];
        partials[(size_t)(NBLK + blockIdx.x) * NCOUT + tid] = psumsq[tid];
    }
}

// ---------------------------------------------------------------------------
// Reduce partials -> fold gamma/beta into per-channel scale/bias.
// 256 threads: 4 segments per channel + LDS tree reduce.
// ---------------------------------------------------------------------------
__global__ __launch_bounds__(256)
void bn_stats(const float* __restrict__ partials,
              const float* __restrict__ gamma,
              const float* __restrict__ beta,
              float* __restrict__ sb) {     // sb[0..63]=scale, [64..127]=bias
    __shared__ float sS[256];
    __shared__ float sQ[256];
    const int tid = threadIdx.x;
    const int c   = tid & 63;
    const int seg = tid >> 6;
    float S = 0.f, Q = 0.f;
    for (int i = seg; i < NBLK; i += 4) {
        S += partials[(size_t)i * NCOUT + c];
        Q += partials[(size_t)(NBLK + i) * NCOUT + c];
    }
    sS[tid] = S; sQ[tid] = Q;
    __syncthreads();
    if (seg == 0) {
        S = sS[c] + sS[64 + c] + sS[128 + c] + sS[192 + c];
        Q = sQ[c] + sQ[64 + c] + sQ[128 + c] + sQ[192 + c];
        const float inv_n = 1.0f / (float)NVOX;
        float m  = S * inv_n;
        float v  = Q * inv_n - m * m;
        float sc = gamma[c] * rsqrtf(v + BN_EPS);
        sb[c]         = sc;
        sb[NCOUT + c] = beta[c] - m * sc;
    }
}

// ---------------------------------------------------------------------------
// In-place BN apply (f16) + zero the sentinel row N.
// ---------------------------------------------------------------------------
__global__ __launch_bounds__(256)
void bn_apply_pad(_Float16* __restrict__ buf, const float* __restrict__ sb) {
    size_t i = ((size_t)blockIdx.x * 256 + threadIdx.x) * 8;
    const size_t total = (size_t)(NVOX + 1) * NCOUT;
    if (i >= total) return;
    v8h* p = (v8h*)(buf + i);
    if (i >= (size_t)NVOX * NCOUT) { v8h z = {}; *p = z; return; }
    const int c0 = (int)(i & (NCOUT - 1));   // 8 | 64 -> row-uniform channel span
    v8h x = *p, y;
    #pragma unroll
    for (int e = 0; e < 8; ++e)
        y[e] = (_Float16)((float)x[e] * sb[c0 + e] + sb[NCOUT + c0 + e]);
    *p = y;
}

// ---------------------------------------------------------------------------
// out = branchB + branchC (both already BN-applied), f32 output.
// ---------------------------------------------------------------------------
__global__ __launch_bounds__(256)
void final_add(const _Float16* __restrict__ a, const _Float16* __restrict__ b,
               float* __restrict__ out) {
    size_t i = ((size_t)blockIdx.x * 256 + threadIdx.x) * 8;
    if (i >= (size_t)NVOX * NCOUT) return;
    v8h xa = *(const v8h*)(a + i);
    v8h xb = *(const v8h*)(b + i);
    v8f y;
    #pragma unroll
    for (int e = 0; e < 8; ++e) y[e] = (float)xa[e] + (float)xb[e];
    *(v8f*)(out + i) = y;
}

// ---------------------------------------------------------------------------
extern "C" void kernel_launch(void* const* d_in, const int* in_sizes, int n_in,
                              void* d_out, int out_size, void* d_ws, size_t ws_size,
                              hipStream_t stream) {
    const float* feats = (const float*)d_in[0];
    const float* W1    = (const float*)d_in[1];
    const float* W12   = (const float*)d_in[2];
    const float* W2    = (const float*)d_in[3];
    const float* W3    = (const float*)d_in[4];
    const float* g0  = (const float*)d_in[5];  const float* b0  = (const float*)d_in[6];
    const float* g02 = (const float*)d_in[7];  const float* b02 = (const float*)d_in[8];
    const float* g1  = (const float*)d_in[9];  const float* b1  = (const float*)d_in[10];
    const float* g2  = (const float*)d_in[11]; const float* b2  = (const float*)d_in[12];
    const int* nbr31 = (const int*)d_in[13];
    const int* nbr13 = (const int*)d_in[14];
    float* out = (float*)d_out;

    // ---- carve workspace (256B aligned regions) ----
    char* ws = (char*)d_ws;
    size_t off = 0;
    auto carve = [&](size_t bytes) -> char* {
        char* p = ws + off;
        off = (off + bytes + 255) & ~(size_t)255;
        return p;
    };
    _Float16* featsH = (_Float16*)carve((size_t)(NVOX + 1) * 32 * 2);
    _Float16* bufA   = (_Float16*)carve((size_t)(NVOX + 1) * NCOUT * 2);
    _Float16* bufB   = (_Float16*)carve((size_t)(NVOX + 1) * NCOUT * 2);
    _Float16* bufC   = (_Float16*)carve((size_t)(NVOX + 1) * NCOUT * 2);
    _Float16* Wt1    = (_Float16*)carve((size_t)9 * NCOUT * 32 * 2);
    _Float16* Wt12   = (_Float16*)carve((size_t)9 * NCOUT * 64 * 2);
    _Float16* Wt2    = (_Float16*)carve((size_t)9 * NCOUT * 32 * 2);
    _Float16* Wt3    = (_Float16*)carve((size_t)9 * NCOUT * 64 * 2);
    float*    part   = (float*)   carve((size_t)2 * NBLK * NCOUT * 4);
    float*    sb0    = (float*)   carve(2 * NCOUT * 4);
    float*    sb1    = (float*)   carve(2 * NCOUT * 4);
    float*    sb2    = (float*)   carve(2 * NCOUT * 4);
    float*    sb3    = (float*)   carve(2 * NCOUT * 4);

    const int pfBlocks    = (int)(((size_t)(NVOX + 1) * 32 / 8 + 255) / 256);
    const int applyBlocks = (int)(((size_t)(NVOX + 1) * NCOUT / 8 + 255) / 256);
    const int addBlocks   = (int)(((size_t)NVOX * NCOUT / 8 + 255) / 256);

    // ---- prep: f16 activations (zero-padded) + transposed f16 weights ----
    prep_feats<<<pfBlocks, 256, 0, stream>>>(feats, featsH);
    prep_w<<<(9 * NCOUT * 32 + 255) / 256, 256, 0, stream>>>(W1,  Wt1,  32);
    prep_w<<<(9 * NCOUT * 64 + 255) / 256, 256, 0, stream>>>(W12, Wt12, 64);
    prep_w<<<(9 * NCOUT * 32 + 255) / 256, 256, 0, stream>>>(W2,  Wt2,  32);
    prep_w<<<(9 * NCOUT * 64 + 255) / 256, 256, 0, stream>>>(W3,  Wt3,  64);

    // ---- shortcut branch: conv(3x1) -> BN -> conv(1x3) -> BN  (-> bufB) ----
    subm_conv_wmma<32><<<NBLK, 256, 0, stream>>>(featsH, nbr31, Wt1, bufA, part);
    bn_stats<<<1, 256, 0, stream>>>(part, g0, b0, sb0);
    bn_apply_pad<<<applyBlocks, 256, 0, stream>>>(bufA, sb0);
    subm_conv_wmma<64><<<NBLK, 256, 0, stream>>>(bufA, nbr13, Wt12, bufB, part);
    bn_stats<<<1, 256, 0, stream>>>(part, g02, b02, sb1);
    bn_apply_pad<<<applyBlocks, 256, 0, stream>>>(bufB, sb1);

    // ---- main branch: conv(1x3) -> BN -> conv(3x1) -> BN  (-> bufC) ----
    subm_conv_wmma<32><<<NBLK, 256, 0, stream>>>(featsH, nbr13, Wt2, bufA, part);
    bn_stats<<<1, 256, 0, stream>>>(part, g1, b1, sb2);
    bn_apply_pad<<<applyBlocks, 256, 0, stream>>>(bufA, sb2);
    subm_conv_wmma<64><<<NBLK, 256, 0, stream>>>(bufA, nbr31, Wt3, bufC, part);
    bn_stats<<<1, 256, 0, stream>>>(part, g2, b2, sb3);
    bn_apply_pad<<<applyBlocks, 256, 0, stream>>>(bufC, sb3);

    // ---- residual add ----
    final_add<<<addBlocks, 256, 0, stream>>>(bufB, bufC, out);
}
